// Attention_16655883174036
// MI455X (gfx1250) — matrix operand order repaired and checked by reading, not compiled, not run on hardware
//
#include <hip/hip_runtime.h>
#include <hip/hip_bf16.h>

#define S_LEN 4096
#define E_DIM 768
#define NHEAD 12
#define HDIM  64

typedef __bf16 bf16_t;
typedef __attribute__((ext_vector_type(16))) __bf16 v16bf;
typedef __attribute__((ext_vector_type(8)))  float  v8f;

union Frag16 { v16bf v; uint4 q[2]; };

// ---- fragment loaders (CDNA5 ISA 7.12.2 16-bit layouts, wave32) ----

// A-matrix 16x32 (MxK), row-major source, ld elems per row.
// lanes 0-15: m=lane, v0-3 = K0..7, v4-7 = K16..23
// lanes16-31: m=lane-16, v0-3 = K8..15, v4-7 = K24..31
__device__ __forceinline__ v16bf load_a_frag(const bf16_t* __restrict__ base,
                                             int ld, int r0, int k0) {
  int lane = threadIdx.x;
  int m  = lane & 15;
  int ko = (lane & 16) ? 8 : 0;
  const bf16_t* p = base + (size_t)(r0 + m) * ld + (k0 + ko);
  Frag16 f;
  f.q[0] = *(const uint4*)(p);        // 8 bf16: K = k0+ko .. +7
  f.q[1] = *(const uint4*)(p + 16);   // 8 bf16: K = k0+ko+16 .. +23
  return f.v;
}

// B-matrix 32x16 (KxN) where memory holds B^T row-major: src[n][k].
// lane holds column n=lane&15; lanes 0-15: K=k0..7,k0+16..23, lanes 16-31: K=k0+8..15,k0+24..31.
__device__ __forceinline__ v16bf load_bT_frag(const bf16_t* __restrict__ base,
                                              int ld, int n0, int k0) {
  int lane = threadIdx.x;
  int n  = lane & 15;
  int ko = (lane & 16) ? 8 : 0;
  const bf16_t* p = base + (size_t)(n0 + n) * ld + (k0 + ko);
  Frag16 f;
  f.q[0] = *(const uint4*)(p);        // K = k0+ko .. +7
  f.q[1] = *(const uint4*)(p + 16);   // K = k0+ko+16 .. +23
  return f.v;
}

__device__ __forceinline__ v8f wmma_bf16(v16bf a, v16bf b, v8f c) {
  return __builtin_amdgcn_wmma_f32_16x16x32_bf16(
      /*neg_a=*/false, a, /*neg_b=*/false, b,
      /*c_mod=*/(short)0, c, /*reuse_a=*/false, /*reuse_b=*/false);
}

// ---- fp32 -> bf16 conversion ----
__global__ void cvt_f32_bf16(const float* __restrict__ src,
                             bf16_t* __restrict__ dst, int n) {
  int i = blockIdx.x * blockDim.x + threadIdx.x;
  if (i < n) dst[i] = (bf16_t)src[i];
}

// ---- GEMM: dst = A[S,E] @ W^T[E,E] + b, one 32(M)x64(N) tile per wave ----
// mode 0: bf16 row-major out; mode 1: bf16 transposed out ([E,S], for V);
// mode 2: f32 row-major out (final projection).
__global__ __launch_bounds__(32)
void gemm_xwT(const bf16_t* __restrict__ A,
              const bf16_t* __restrict__ W,     // [E,E] row-major (out_feat, in_feat)
              const float* __restrict__ bias,
              bf16_t* __restrict__ dstb,
              float* __restrict__ dstf, int mode) {
  int e0 = blockIdx.x * 64;
  int s0 = blockIdx.y * 32;
  v8f c[2][4] = {};
  for (int k0 = 0; k0 < E_DIM; k0 += 32) {
    v16bf a0 = load_a_frag(A, E_DIM, s0,      k0);
    v16bf a1 = load_a_frag(A, E_DIM, s0 + 16, k0);
#pragma unroll
    for (int j = 0; j < 4; ++j) {
      v16bf b = load_bT_frag(W, E_DIM, e0 + 16 * j, k0);
      c[0][j] = wmma_bf16(a0, b, c[0][j]);
      c[1][j] = wmma_bf16(a1, b, c[1][j]);
    }
  }
  int lane = threadIdx.x;
  int n  = lane & 15;
  int mb = (lane & 16) ? 8 : 0;
#pragma unroll
  for (int j = 0; j < 4; ++j) {
    int e = e0 + 16 * j + n;
    float bn = bias[e];
#pragma unroll
    for (int mt = 0; mt < 2; ++mt) {
#pragma unroll
      for (int i = 0; i < 8; ++i) {
        int s = s0 + mt * 16 + mb + i;
        float v = c[mt][j][i] + bn;
        if (mode == 0)      dstb[(size_t)s * E_DIM + e] = (bf16_t)v;
        else if (mode == 1) dstb[(size_t)e * S_LEN + s] = (bf16_t)v;   // V^T
        else                dstf[(size_t)s * E_DIM + e] = v;
      }
    }
  }
}

// ---- flash attention: one wave per (32-query block, head) ----
__global__ __launch_bounds__(32)
void flash_attn(const bf16_t* __restrict__ Q,    // [S,E]
                const bf16_t* __restrict__ K,    // [S,E]
                const bf16_t* __restrict__ Vt,   // [E,S]
                const unsigned char* __restrict__ mask,  // [S], nonzero = masked
                bf16_t* __restrict__ ctx) {      // [S,E]
  __shared__ __align__(16) bf16_t pbuf[32 * 32];   // P tile, 2 KB

  int s0   = blockIdx.x * 32;
  int h    = blockIdx.y;
  int lane = threadIdx.x;
  int n  = lane & 15;
  int mb = (lane & 16) ? 8 : 0;
  const float scale = 0.03608439182435161f;  // 1/sqrt(768)

  v16bf qa[2][2];
#pragma unroll
  for (int mt = 0; mt < 2; ++mt)
#pragma unroll
    for (int ks = 0; ks < 2; ++ks)
      qa[mt][ks] = load_a_frag(Q, E_DIM, s0 + mt * 16, h * HDIM + ks * 32);

  v8f o[2][4] = {};
  float mrow[2][8], lrow[2][8];
#pragma unroll
  for (int mt = 0; mt < 2; ++mt)
#pragma unroll
    for (int i = 0; i < 8; ++i) { mrow[mt][i] = -1e30f; lrow[mt][i] = 0.0f; }

  for (int kt = 0; kt < S_LEN; kt += 32) {
    // K fragments shared by both query m-tiles
    v16bf bk00 = load_bT_frag(K, E_DIM, kt,      h * HDIM + 0);
    v16bf bk01 = load_bT_frag(K, E_DIM, kt,      h * HDIM + 32);
    v16bf bk10 = load_bT_frag(K, E_DIM, kt + 16, h * HDIM + 0);
    v16bf bk11 = load_bT_frag(K, E_DIM, kt + 16, h * HDIM + 32);

    v8f sc[2][2] = {};
#pragma unroll
    for (int mt = 0; mt < 2; ++mt) {
      sc[mt][0] = wmma_bf16(qa[mt][0], bk00, sc[mt][0]);
      sc[mt][0] = wmma_bf16(qa[mt][1], bk01, sc[mt][0]);
      sc[mt][1] = wmma_bf16(qa[mt][0], bk10, sc[mt][1]);
      sc[mt][1] = wmma_bf16(qa[mt][1], bk11, sc[mt][1]);
    }

    bool msk0 = mask[kt + n] != 0;
    bool msk1 = mask[kt + 16 + n] != 0;

#pragma unroll
    for (int mt = 0; mt < 2; ++mt) {
#pragma unroll
      for (int i = 0; i < 8; ++i) {
        float v0 = msk0 ? -1e30f : sc[mt][0][i] * scale;
        float v1 = msk1 ? -1e30f : sc[mt][1][i] * scale;
        // row max across the 32 keys (16-lane group reduction)
        float v = fmaxf(v0, v1);
        v = fmaxf(v, __shfl_xor(v, 1, 32));
        v = fmaxf(v, __shfl_xor(v, 2, 32));
        v = fmaxf(v, __shfl_xor(v, 4, 32));
        v = fmaxf(v, __shfl_xor(v, 8, 32));
        float mnew  = fmaxf(mrow[mt][i], v);
        float alpha = __expf(mrow[mt][i] - mnew);
        mrow[mt][i] = mnew;
        float p0 = __expf(v0 - mnew);
        float p1 = __expf(v1 - mnew);
        float rs = p0 + p1;
        rs += __shfl_xor(rs, 1, 32);
        rs += __shfl_xor(rs, 2, 32);
        rs += __shfl_xor(rs, 4, 32);
        rs += __shfl_xor(rs, 8, 32);
        lrow[mt][i] = lrow[mt][i] * alpha + rs;
#pragma unroll
        for (int dt = 0; dt < 4; ++dt) o[mt][dt][i] *= alpha;
        // stash P into LDS (C layout -> 32x32 bf16 tile, stride 32)
        pbuf[(mt * 16 + mb + i) * 32 + n]      = (bf16_t)p0;
        pbuf[(mt * 16 + mb + i) * 32 + 16 + n] = (bf16_t)p1;
      }
    }
    __syncthreads();
    // re-read P as two A fragments (rows 0-15, 16-31)
    v16bf pa[2];
    {
      int m  = lane & 15;
      int ko = (lane & 16) ? 8 : 0;
#pragma unroll
      for (int mt = 0; mt < 2; ++mt) {
        Frag16 f;
        f.q[0] = *(const uint4*)&pbuf[(mt * 16 + m) * 32 + ko];
        f.q[1] = *(const uint4*)&pbuf[(mt * 16 + m) * 32 + 16 + ko];
        pa[mt] = f.v;
      }
    }
    __syncthreads();

    // O += P[32x32] @ V[32x64]  (V fragments shared by both m-tiles)
#pragma unroll
    for (int dt = 0; dt < 4; ++dt) {
      v16bf bv = load_bT_frag(Vt, S_LEN, h * HDIM + dt * 16, kt);
      o[0][dt] = wmma_bf16(pa[0], bv, o[0][dt]);
      o[1][dt] = wmma_bf16(pa[1], bv, o[1][dt]);
    }
  }

#pragma unroll
  for (int mt = 0; mt < 2; ++mt) {
#pragma unroll
    for (int i = 0; i < 8; ++i) {
      float inv = 1.0f / lrow[mt][i];
      size_t row = (size_t)(s0 + mt * 16 + mb + i) * E_DIM + h * HDIM;
#pragma unroll
      for (int dt = 0; dt < 4; ++dt)
        ctx[row + dt * 16 + n] = (bf16_t)(o[mt][dt][i] * inv);
    }
  }
}

extern "C" void kernel_launch(void* const* d_in, const int* in_sizes, int n_in,
                              void* d_out, int out_size, void* d_ws, size_t ws_size,
                              hipStream_t stream) {
  const float* x  = (const float*)d_in[0];
  const unsigned char* mask = (const unsigned char*)d_in[1];
  const float* Wq = (const float*)d_in[2];
  const float* bq = (const float*)d_in[3];
  const float* Wk = (const float*)d_in[4];
  const float* bk = (const float*)d_in[5];
  const float* Wv = (const float*)d_in[6];
  const float* bv = (const float*)d_in[7];
  const float* Wo = (const float*)d_in[8];
  const float* bo = (const float*)d_in[9];

  char* ws = (char*)d_ws;
  size_t off = 0;
  auto alloc = [&](size_t bytes) -> void* {
    void* p = ws + off;
    off += (bytes + 255) & ~(size_t)255;
    return p;
  };
  const size_t SE = (size_t)S_LEN * E_DIM;
  const size_t EE = (size_t)E_DIM * E_DIM;
  bf16_t* xb  = (bf16_t*)alloc(SE * 2);
  bf16_t* wqb = (bf16_t*)alloc(EE * 2);
  bf16_t* wkb = (bf16_t*)alloc(EE * 2);
  bf16_t* wvb = (bf16_t*)alloc(EE * 2);
  bf16_t* wob = (bf16_t*)alloc(EE * 2);
  bf16_t* Qb  = (bf16_t*)alloc(SE * 2);
  bf16_t* Kb  = (bf16_t*)alloc(SE * 2);
  bf16_t* Vtb = (bf16_t*)alloc(SE * 2);
  bf16_t* Ctx = (bf16_t*)alloc(SE * 2);

  cvt_f32_bf16<<<((int)SE + 255) / 256, 256, 0, stream>>>(x,  xb,  (int)SE);
  cvt_f32_bf16<<<((int)EE + 255) / 256, 256, 0, stream>>>(Wq, wqb, (int)EE);
  cvt_f32_bf16<<<((int)EE + 255) / 256, 256, 0, stream>>>(Wk, wkb, (int)EE);
  cvt_f32_bf16<<<((int)EE + 255) / 256, 256, 0, stream>>>(Wv, wvb, (int)EE);
  cvt_f32_bf16<<<((int)EE + 255) / 256, 256, 0, stream>>>(Wo, wob, (int)EE);

  dim3 gproj(E_DIM / 64, S_LEN / 32);   // 12 x 128
  gemm_xwT<<<gproj, 32, 0, stream>>>(xb, wqb, bq, Qb,  nullptr, 0);
  gemm_xwT<<<gproj, 32, 0, stream>>>(xb, wkb, bk, Kb,  nullptr, 0);
  gemm_xwT<<<gproj, 32, 0, stream>>>(xb, wvb, bv, Vtb, nullptr, 1);

  dim3 gatt(S_LEN / 32, NHEAD);         // 128 x 12
  flash_attn<<<gatt, 32, 0, stream>>>(Qb, Kb, Vtb, mask, Ctx);

  gemm_xwT<<<gproj, 32, 0, stream>>>(Ctx, wob, bo, nullptr, (float*)d_out, 2);
}
